// Text_Graph_Encoder_43413529428591
// MI455X (gfx1250) — matrix operand order repaired and checked by reading, not compiled, not run on hardware
//
#include <hip/hip_runtime.h>
#include <hip/hip_bf16.h>
#include <math.h>

// ---------------- problem constants ----------------
constexpr int B  = 64;
constexpr int N  = 1024;
constexpr int E  = 16384;
constexpr int D  = 300;
constexpr int H  = 5;
constexpr int L  = 2;
constexpr int HD = H * D;          // 1500
constexpr int KP = 320;            // K padded to 10 * 32
constexpr int NCP = 1536;          // HD padded to 24 * 64
constexpr int NK = KP / 32;        // 10 K-steps
constexpr float NEG_SLOPE = 0.2f;
constexpr float LN_EPS = 1e-5f;

typedef __attribute__((ext_vector_type(16))) __bf16 v16bf;
typedef __attribute__((ext_vector_type(8)))  __bf16 bf16x8;
typedef __attribute__((ext_vector_type(8)))  float  v8f;
typedef __attribute__((ext_vector_type(4)))  int    v4i;

typedef __attribute__((address_space(1))) v4i g_v4i;   // global b128
typedef __attribute__((address_space(3))) v4i l_v4i;   // LDS b128

// ---------------- small helpers ----------------
__device__ inline float warp_sum32(float v) {
    #pragma unroll
    for (int o = 16; o > 0; o >>= 1) v += __shfl_xor(v, o, 32);
    return v;
}
__device__ inline unsigned enc_f32(float f) {
    unsigned u = __float_as_uint(f);
    return (u & 0x80000000u) ? ~u : (u | 0x80000000u);
}
__device__ inline float dec_f32(unsigned u) {
    unsigned v = (u & 0x80000000u) ? (u & 0x7fffffffu) : ~u;
    return __uint_as_float(v);
}
__device__ inline float gelu_tanh(float x) {
    float x3 = x * x * x;
    return 0.5f * x * (1.0f + tanhf(0.7978845608028654f * (x + 0.044715f * x3)));
}

// ---- CDNA5 async global->LDS copy (ASYNCcnt-tracked), 16B per lane ----
__device__ inline void async_copy_b128(const __bf16* g, __bf16* l) {
#if __has_builtin(__builtin_amdgcn_global_load_async_to_lds_b128)
    __builtin_amdgcn_global_load_async_to_lds_b128(
        (g_v4i*)g, (l_v4i*)l, 0, 0);
#else
    unsigned loff = (unsigned)(unsigned long long)(
        (__attribute__((address_space(3))) void*)l);
    asm volatile("global_load_async_to_lds_b128 %0, %1, off"
                 :: "v"(loff), "v"(g) : "memory");
#endif
}
template <int NCNT>
__device__ inline void wait_asynccnt() {
#if __has_builtin(__builtin_amdgcn_s_wait_asynccnt)
    __builtin_amdgcn_s_wait_asynccnt(NCNT);
#else
    asm volatile("s_wait_asynccnt %0" :: "i"(NCNT) : "memory");
#endif
}

// ---------------- conversion / init kernels ----------------
// W [L, 300, 1500] f32 -> WbfT [L, 1536, 320] bf16 (transposed, zero padded)
__global__ void k_convert_w(const float* __restrict__ W, __bf16* __restrict__ WbfT) {
    int idx = blockIdx.x * blockDim.x + threadIdx.x;
    int total = L * NCP * KP;
    if (idx >= total) return;
    int k = idx % KP;
    int n = (idx / KP) % NCP;
    int l = idx / (KP * NCP);
    float v = 0.f;
    if (k < D && n < HD) v = W[((size_t)l * D + k) * HD + n];
    WbfT[idx] = (__bf16)v;
}

// x [B, N, 300] f32 -> Xbf [B, N, 320] bf16, zero padded
__global__ void k_convert_x(const float* __restrict__ xin, __bf16* __restrict__ Xbf) {
    int idx = blockIdx.x * blockDim.x + threadIdx.x;
    int total = B * N * KP;
    if (idx >= total) return;
    int c = idx % KP;
    int row = idx / KP;   // b*N + n
    float v = 0.f;
    if (c < D) v = xin[(size_t)row * D + c];
    Xbf[idx] = (__bf16)v;
}

__global__ void k_zero2(unsigned* __restrict__ mEnc, float* __restrict__ den) {
    int idx = blockIdx.x * blockDim.x + threadIdx.x;
    if (idx >= B * N * H) return;
    mEnc[idx] = 0u;          // decodes below every real score encoding
    den[idx]  = 0.f;
}

__global__ void k_init_acc(const float* __restrict__ bias_l, float* __restrict__ Acc) {
    int idx = blockIdx.x * blockDim.x + threadIdx.x;
    if (idx >= B * N * D) return;
    Acc[idx] = bias_l[idx % D];
}

// ---------------- WMMA projection GEMM ----------------
// Hproj[b] = Xbf[b] (1024 x 320) @ W_l (320 x 1536, stored transposed) -> fp32
// block = 256 threads = 8 waves; block tile 128(M) x 64(N); wave -> 32x32 tile
// (4 wmma / K-step / wave); LDS double-buffered, filled with async b128 copies.
constexpr int A_ELEMS = 128 * 40;            // 32-K rows padded to 40
constexpr int B_ELEMS = 64 * 40;
constexpr int BUF_ELEMS = A_ELEMS + B_ELEMS; // 7680 bf16 per buffer

__device__ inline void stage_tiles(const __bf16* __restrict__ Xg,
                                   const __bf16* __restrict__ WT,
                                   __bf16* __restrict__ bufA,
                                   __bf16* __restrict__ bufB,
                                   int m0, int n0, int k0, int tid) {
    // A tile: 128 rows x 32 K, 2 x 16B per thread
    {
        int r0 = tid >> 2,          c0 = (tid & 3) * 8;
        int r1 = (tid + 256) >> 2,  c1 = (tid & 3) * 8;
        async_copy_b128(Xg + (size_t)(m0 + r0) * KP + k0 + c0, bufA + r0 * 40 + c0);
        async_copy_b128(Xg + (size_t)(m0 + r1) * KP + k0 + c1, bufA + r1 * 40 + c1);
    }
    // B tile (transposed W): 64 cols x 32 K, 1 x 16B per thread
    {
        int r = tid >> 2, c = (tid & 3) * 8;
        async_copy_b128(WT + (size_t)(n0 + r) * KP + k0 + c, bufB + r * 40 + c);
    }
}

__global__ void __launch_bounds__(256)
k_gemm(const __bf16* __restrict__ Xbf, const __bf16* __restrict__ WT_l,
       float* __restrict__ Hproj) {
    __shared__ __align__(16) __bf16 smem[2 * BUF_ELEMS];

    const int tid  = threadIdx.x;
    const int lane = tid & 31;
    const int w    = tid >> 5;       // 0..7
    const int wm   = w & 3;          // M sub-tile (32 rows)
    const int wn   = w >> 2;         // N sub-tile (32 cols)

    const int b  = blockIdx.z;
    const int m0 = blockIdx.y * 128;
    const int n0 = blockIdx.x * 64;

    const __bf16* Xg = Xbf + (size_t)b * N * KP;

    v8f acc[2][2] = {{{}, {}}, {{}, {}}};

    const int halfsel = lane >> 4;          // 0: lanes 0-15, 1: lanes 16-31
    const int kbaseA  = halfsel * 8;        // A frag: K 0..7 / 8..15 (+16 upper half)
    const int kbaseB  = halfsel * 8;        // B frag: K 0..7 / 8..15 (+16 upper half)
    const int l15     = lane & 15;

    stage_tiles(Xg, WT_l, smem, smem + A_ELEMS, m0, n0, 0, tid);

    #pragma unroll 1
    for (int kt = 0; kt < NK; ++kt) {
        const int cur = kt & 1;
        __bf16* As  = smem + cur * BUF_ELEMS;
        __bf16* BsT = As + A_ELEMS;

        if (kt + 1 < NK) {
            __bf16* nbuf = smem + (cur ^ 1) * BUF_ELEMS;
            stage_tiles(Xg, WT_l, nbuf, nbuf + A_ELEMS, m0, n0, (kt + 1) * 32, tid);
            wait_asynccnt<3>();   // this wave's 3 current-tile copies retired (in order)
        } else {
            wait_asynccnt<0>();
        }
        __syncthreads();

        // fragments (ISA 7.12.2 16-bit A 16x32 / B 32x16 lane layouts)
        v16bf afrag[2], bfrag[2];
        #pragma unroll
        for (int s = 0; s < 2; ++s) {
            int mrow = wm * 32 + s * 16 + l15;
            bf16x8 alo = *(const bf16x8*)&As[mrow * 40 + kbaseA];
            bf16x8 ahi = *(const bf16x8*)&As[mrow * 40 + kbaseA + 16];
            afrag[s] = __builtin_shufflevector(alo, ahi,
                0,1,2,3,4,5,6,7,8,9,10,11,12,13,14,15);
            int col = wn * 32 + s * 16 + l15;
            bf16x8 blo = *(const bf16x8*)&BsT[col * 40 + kbaseB];
            bf16x8 bhi = *(const bf16x8*)&BsT[col * 40 + kbaseB + 16];
            bfrag[s] = __builtin_shufflevector(blo, bhi,
                0,1,2,3,4,5,6,7,8,9,10,11,12,13,14,15);
        }
        #pragma unroll
        for (int ms = 0; ms < 2; ++ms)
            #pragma unroll
            for (int ns = 0; ns < 2; ++ns)
                acc[ms][ns] = __builtin_amdgcn_wmma_f32_16x16x32_bf16(
                    false, afrag[ms], false, bfrag[ns], (short)0,
                    acc[ms][ns], false, false);

        __syncthreads();
    }

    // store: C/D layout — lane<16: M=r, lane>=16: M=8+r; N = lane&15
    float* Og = Hproj + (size_t)b * N * HD;
    #pragma unroll
    for (int ns = 0; ns < 2; ++ns) {
        const int ocol = n0 + wn * 32 + ns * 16 + l15;
        if (ocol >= HD) continue;
        #pragma unroll
        for (int ms = 0; ms < 2; ++ms) {
            const int orow0 = m0 + wm * 32 + ms * 16 + halfsel * 8;
            #pragma unroll
            for (int r = 0; r < 8; ++r)
                Og[(size_t)(orow0 + r) * HD + ocol] = acc[ms][ns][r];
        }
    }
}

// ---------------- attention coefficients ----------------
__global__ void __launch_bounds__(256)
k_attn(const float* __restrict__ Hproj, const float* __restrict__ asrc_l,
       const float* __restrict__ adst_l, float* __restrict__ a_s,
       float* __restrict__ a_d) {
    int widx = blockIdx.x * 8 + (threadIdx.x >> 5);
    if (widx >= B * N * H) return;
    int lane = threadIdx.x & 31;
    int h   = widx % H;
    int bn  = widx / H;
    const float* hp = Hproj + (size_t)bn * HD + h * D;
    const float* vs = asrc_l + h * D;
    const float* vd = adst_l + h * D;
    float ss = 0.f, sd = 0.f;
    for (int d = lane; d < D; d += 32) {
        float x = hp[d];
        ss += x * vs[d];
        sd += x * vd[d];
    }
    ss = warp_sum32(ss);
    sd = warp_sum32(sd);
    if (lane == 0) { a_s[widx] = ss; a_d[widx] = sd; }
}

// ---------------- edge passes ----------------
__device__ inline float edge_score(const int* ei, const unsigned char* mask,
                                   const float* a_s, const float* a_d,
                                   int b, int e, int h, int& dst_out) {
    int src = ei[(size_t)b * 2 * E + e];
    int dst = ei[(size_t)b * 2 * E + E + e];
    dst_out = dst;
    float s = a_s[((size_t)b * N + src) * H + h] + a_d[((size_t)b * N + dst) * H + h];
    s = (s > 0.f) ? s : NEG_SLOPE * s;             // leaky_relu
    if (mask[(size_t)b * N + src]) s = -1e9f;      // drop padded-source messages
    return s;
}

__global__ void k_edge_max(const int* __restrict__ ei, const unsigned char* __restrict__ mask,
                           const float* __restrict__ a_s, const float* __restrict__ a_d,
                           unsigned* __restrict__ mEnc) {
    int idx = blockIdx.x * blockDim.x + threadIdx.x;
    if (idx >= B * E * H) return;
    int h = idx % H;
    int e = (idx / H) % E;
    int b = idx / (E * H);
    int dst;
    float s = edge_score(ei, mask, a_s, a_d, b, e, h, dst);
    atomicMax(&mEnc[((size_t)b * N + dst) * H + h], enc_f32(s));
}

__global__ void k_edge_exp(const int* __restrict__ ei, const unsigned char* __restrict__ mask,
                           const float* __restrict__ a_s, const float* __restrict__ a_d,
                           const unsigned* __restrict__ mEnc, float* __restrict__ den,
                           float* __restrict__ Ex) {
    int idx = blockIdx.x * blockDim.x + threadIdx.x;
    if (idx >= B * E * H) return;
    int h = idx % H;
    int e = (idx / H) % E;
    int b = idx / (E * H);
    int dst;
    float s = edge_score(ei, mask, a_s, a_d, b, e, h, dst);
    float m = dec_f32(mEnc[((size_t)b * N + dst) * H + h]);
    float ex = __expf(s - m);
    Ex[((size_t)b * E + e) * H + h] = ex;
    atomicAdd(&den[((size_t)b * N + dst) * H + h], ex);
}

// ---------------- aggregation (one wave per edge) ----------------
__global__ void __launch_bounds__(256)
k_aggregate(const int* __restrict__ ei, const float* __restrict__ Ex,
            const float* __restrict__ den, const float* __restrict__ Hproj,
            float* __restrict__ Acc) {
    int widx = blockIdx.x * 8 + (threadIdx.x >> 5);
    if (widx >= B * E) return;
    int lane = threadIdx.x & 31;
    int e = widx % E;
    int b = widx / E;
    int src = ei[(size_t)b * 2 * E + e];
    int dst = ei[(size_t)b * 2 * E + E + e];

    float wgt = 0.f;
    if (lane < H) {
        float dn = den[((size_t)b * N + dst) * H + lane] + 1e-16f;
        wgt = Ex[((size_t)b * E + e) * H + lane] / dn * (1.0f / H);  // alpha/H (head mean)
    }
    float w0 = __shfl(wgt, 0, 32), w1 = __shfl(wgt, 1, 32), w2 = __shfl(wgt, 2, 32),
          w3 = __shfl(wgt, 3, 32), w4 = __shfl(wgt, 4, 32);

    const float* hp = Hproj + ((size_t)b * N + src) * HD;
    float* op = Acc + ((size_t)b * N + dst) * D;
    for (int d = lane; d < D; d += 32) {
        float acc = w0 * hp[0 * D + d] + w1 * hp[1 * D + d] + w2 * hp[2 * D + d]
                  + w3 * hp[3 * D + d] + w4 * hp[4 * D + d];
        atomicAdd(&op[d], acc);
    }
}

// ---------------- fused GELU + LayerNorm (one wave per node) ----------------
__global__ void __launch_bounds__(256)
k_epilogue(const float* __restrict__ Acc, const float* __restrict__ gamma,
           const float* __restrict__ beta, float* __restrict__ outp) {
    int widx = blockIdx.x * 8 + (threadIdx.x >> 5);
    if (widx >= B * N) return;
    int lane = threadIdx.x & 31;
    const float* xin = Acc + (size_t)widx * D;
    float g[10];
    float s = 0.f;
    #pragma unroll
    for (int i = 0; i < 10; ++i) {
        int d = lane + i * 32;
        float t = 0.f;
        if (d < D) t = gelu_tanh(xin[d]);
        g[i] = t;
        s += t;
    }
    s = warp_sum32(s);
    float mu = s * (1.0f / D);
    float v = 0.f;
    #pragma unroll
    for (int i = 0; i < 10; ++i) {
        int d = lane + i * 32;
        if (d < D) { float dd = g[i] - mu; v += dd * dd; }
    }
    v = warp_sum32(v);
    float rs = rsqrtf(v * (1.0f / D) + LN_EPS);
    float* op = outp + (size_t)widx * D;
    #pragma unroll
    for (int i = 0; i < 10; ++i) {
        int d = lane + i * 32;
        if (d < D) op[d] = (g[i] - mu) * rs * gamma[d] + beta[d];
    }
}

// ---------------- launch ----------------
extern "C" void kernel_launch(void* const* d_in, const int* in_sizes, int n_in,
                              void* d_out, int out_size, void* d_ws, size_t ws_size,
                              hipStream_t stream) {
    const float* t2        = (const float*)d_in[0];
    const int*   edge_idx  = (const int*)d_in[1];
    const unsigned char* gnn_mask = (const unsigned char*)d_in[2];
    const float* W         = (const float*)d_in[3];
    const float* att_src   = (const float*)d_in[4];
    const float* att_dst   = (const float*)d_in[5];
    const float* bias      = (const float*)d_in[6];
    const float* ln_g      = (const float*)d_in[7];
    const float* ln_b      = (const float*)d_in[8];
    float* out = (float*)d_out;

    // workspace carve (256B aligned slices)
    size_t off = 0;
    auto carve = [&](size_t bytes) -> void* {
        void* p = (char*)d_ws + off;
        off += (bytes + 255) & ~(size_t)255;
        return p;
    };
    __bf16*   Xbf   = (__bf16*)  carve((size_t)B * N * KP * sizeof(__bf16));
    __bf16*   WbfT  = (__bf16*)  carve((size_t)L * NCP * KP * sizeof(__bf16));
    float*    Hproj = (float*)   carve((size_t)B * N * HD * sizeof(float));
    float*    a_s   = (float*)   carve((size_t)B * N * H * sizeof(float));
    float*    a_d   = (float*)   carve((size_t)B * N * H * sizeof(float));
    unsigned* mEnc  = (unsigned*)carve((size_t)B * N * H * sizeof(unsigned));
    float*    den   = (float*)   carve((size_t)B * N * H * sizeof(float));
    float*    Ex    = (float*)   carve((size_t)B * E * H * sizeof(float));
    float*    Acc   = (float*)   carve((size_t)B * N * D * sizeof(float));
    float*    Xcur  = (float*)   carve((size_t)B * N * D * sizeof(float));

    {   // one-time transposed weight conversion (both layers)
        int total = L * NCP * KP;
        k_convert_w<<<(total + 255) / 256, 256, 0, stream>>>(W, WbfT);
    }

    for (int l = 0; l < L; ++l) {
        const float* xin  = (l == 0) ? t2 : Xcur;
        float*       outp = (l == L - 1) ? out : Xcur;

        { int total = B * N * KP;
          k_convert_x<<<(total + 255) / 256, 256, 0, stream>>>(xin, Xbf); }

        dim3 ggrid(NCP / 64, N / 128, B);
        k_gemm<<<ggrid, 256, 0, stream>>>(Xbf, WbfT + (size_t)l * NCP * KP, Hproj);

        { int waves = B * N * H;
          k_attn<<<(waves + 7) / 8, 256, 0, stream>>>(
              Hproj, att_src + (size_t)l * H * D, att_dst + (size_t)l * H * D, a_s, a_d); }

        { int total = B * N * H;
          k_zero2<<<(total + 255) / 256, 256, 0, stream>>>(mEnc, den); }

        { int total = B * E * H;
          k_edge_max<<<(total + 255) / 256, 256, 0, stream>>>(
              edge_idx, gnn_mask, a_s, a_d, mEnc);
          k_edge_exp<<<(total + 255) / 256, 256, 0, stream>>>(
              edge_idx, gnn_mask, a_s, a_d, mEnc, den, Ex); }

        { int total = B * N * D;
          k_init_acc<<<(total + 255) / 256, 256, 0, stream>>>(bias + (size_t)l * D, Acc); }

        { int waves = B * E;
          k_aggregate<<<(waves + 7) / 8, 256, 0, stream>>>(edge_idx, Ex, den, Hproj, Acc); }

        { int waves = B * N;
          k_epilogue<<<(waves + 7) / 8, 256, 0, stream>>>(Acc, ln_g, ln_b, outp); }
    }
}
